// KnowledgeStorage_78391743087162
// MI455X (gfx1250) — compile-verified, matrix operand order checked by reading, not run on hardware
//
#include <hip/hip_runtime.h>
#include <hip/hip_bf16.h>
#include <math.h>

// KnowledgeStorage graph module for MI455X (gfx1250).
//  * Cross-edge concat linears factored algebraically (~350 -> ~120 GFLOP).
//  * All GEMMs: v_wmma_f32_16x16x32_bf16, block tile 128x128x32, 8 waves,
//    64x32 per wave (8 WMMA / K-step), double-buffered LDS with
//    register-staged global fetches held in scalar uint4s (no alloca ->
//    no scratch spills), full-tile / edge-tile split so the hot path has
//    no EXEC-mask guards.
//  * Softmax / gating / weighted sums in f32 elementwise kernels.
//  * Scratch bump-allocated from d_ws; every launch on `stream`.

typedef __bf16 bf16_t;
typedef __attribute__((ext_vector_type(16))) __bf16 v16bf;
typedef __attribute__((ext_vector_type(8)))  float  v8f;

#define B_ 8
#define R_ 10
#define O_ 36
#define L_ 512
#define V_ 2048

static __device__ __forceinline__ bf16_t f2bf(float f) {
  unsigned u = __builtin_bit_cast(unsigned, f);
  u = (u + 0x7FFFu + ((u >> 16) & 1u)) >> 16;   // round-to-nearest-even
  unsigned short h = (unsigned short)u;
  return __builtin_bit_cast(bf16_t, h);
}

// ---------------------------------------------------------------------------
// WMMA GEMM: C[M,N] = A[M,K] (bf16) @ W[N,K]^T (bf16) (+ bias[N])
// ---------------------------------------------------------------------------
#define GBM 128
#define GBN 128
#define GBK 32
#define GPAD 8                 // row = 40 bf16 = 80 B (multiple of 16 -> b128 ok)
#define LDROW (GBK + GPAD)

static __device__ __forceinline__ v16bf ld_frag(const bf16_t* p, int kh) {
  // fragment = two contiguous 16B chunks: k = kh+0..7 and 16+kh+0..7
  union { v16bf v; uint4 q[2]; } u;
  u.q[0] = *(const uint4*)(p + kh);
  u.q[1] = *(const uint4*)(p + 16 + kh);
  return u.v;
}

// one 16B chunk of a 128x32 bf16 tile; idx in [0,512)
static __device__ __forceinline__ uint4 fetch_chunk(
    const bf16_t* __restrict__ G, int ld, int idx, int valid, bool guard) {
  int row = idx >> 2;
  int ch  = (idx & 3) * 8;
  uint4 v = {0u, 0u, 0u, 0u};
  if (!guard || row < valid) v = *(const uint4*)(G + (size_t)row * ld + ch);
  return v;
}

static __device__ __forceinline__ void stash_chunk(
    bf16_t (*S)[LDROW], int idx, uint4 v) {
  int row = idx >> 2;
  int ch  = (idx & 3) * 8;
  *(uint4*)(&S[row][ch]) = v;
}

template <bool EDGE>
__global__ void __launch_bounds__(256) k_gemm_wmma(
    const bf16_t* __restrict__ A, int lda,
    const bf16_t* __restrict__ W, int ldw,
    const float* __restrict__ bias,
    float* __restrict__ C, int ldc,
    int M, int N, int K, int m_start) {
  __shared__ bf16_t As[2][GBM][LDROW];   // 2 x 10 KB
  __shared__ bf16_t Bs[2][GBN][LDROW];   // 2 x 10 KB

  const int tid  = threadIdx.x;
  const int lane = tid & 31;
  const int wid  = tid >> 5;           // 8 waves
  const int wm   = wid >> 2;           // 0..1 -> 64-row slab
  const int wn   = wid & 3;            // 0..3 -> 32-col slab
  const int m0   = m_start + blockIdx.y * GBM;
  const int n0   = blockIdx.x * GBN;   // N is a multiple of 128 in all uses
  const int mv   = EDGE ? (M - m0) : GBM;

  v8f acc[4][2];
  #pragma unroll
  for (int i = 0; i < 4; ++i)
    #pragma unroll
    for (int j = 0; j < 2; ++j)
      #pragma unroll
      for (int e = 0; e < 8; ++e) acc[i][j][e] = 0.0f;

  const int kh = (lane >> 4) * 8;      // lane-half k offset (A & B layouts)
  const int rl = lane & 15;            // row (A) / col (B) within 16-tile

  const bf16_t* Abase = A + (size_t)m0 * lda;
  const bf16_t* Wbase = W + (size_t)n0 * ldw;

  // register-staged tile chunks (scalars -> guaranteed VGPRs, no alloca)
  uint4 a0 = fetch_chunk(Abase, lda, tid,       mv,  EDGE);
  uint4 a1 = fetch_chunk(Abase, lda, tid + 256, mv,  EDGE);
  uint4 b0 = fetch_chunk(Wbase, ldw, tid,       GBN, false);
  uint4 b1 = fetch_chunk(Wbase, ldw, tid + 256, GBN, false);
  stash_chunk(As[0], tid,       a0);
  stash_chunk(As[0], tid + 256, a1);
  stash_chunk(Bs[0], tid,       b0);
  stash_chunk(Bs[0], tid + 256, b1);
  __syncthreads();

  int cur = 0;
  for (int k0 = 0; k0 < K; k0 += GBK) {
    const bool hasNext = (k0 + GBK < K);
    if (hasNext) {
      // issue next-tile global loads early; LDS stores happen after the WMMAs
      a0 = fetch_chunk(Abase + k0 + GBK, lda, tid,       mv,  EDGE);
      a1 = fetch_chunk(Abase + k0 + GBK, lda, tid + 256, mv,  EDGE);
      b0 = fetch_chunk(Wbase + k0 + GBK, ldw, tid,       GBN, false);
      b1 = fetch_chunk(Wbase + k0 + GBK, ldw, tid + 256, GBN, false);
      if (k0 + 2 * GBK < K) {
        int pr = EDGE ? (((tid >> 1) < mv) ? (tid >> 1) : 0) : (tid >> 1);
        __builtin_prefetch(Abase + (size_t)pr * lda + k0 + 2 * GBK, 0, 1);
        __builtin_prefetch(Wbase + (size_t)(tid >> 1) * ldw + k0 + 2 * GBK, 0, 1);
      }
    }

    v16bf af[4], bfr[2];
    #pragma unroll
    for (int i = 0; i < 4; ++i)
      af[i] = ld_frag(&As[cur][wm * 64 + i * 16 + rl][0], kh);
    #pragma unroll
    for (int j = 0; j < 2; ++j)
      bfr[j] = ld_frag(&Bs[cur][wn * 32 + j * 16 + rl][0], kh);

    #pragma unroll
    for (int i = 0; i < 4; ++i)
      #pragma unroll
      for (int j = 0; j < 2; ++j)
        acc[i][j] = __builtin_amdgcn_wmma_f32_16x16x32_bf16(
            false, af[i], false, bfr[j], (short)0, acc[i][j], false, false);

    if (hasNext) {
      stash_chunk(As[cur ^ 1], tid,       a0);
      stash_chunk(As[cur ^ 1], tid + 256, a1);
      stash_chunk(Bs[cur ^ 1], tid,       b0);
      stash_chunk(Bs[cur ^ 1], tid + 256, b1);
    }
    __syncthreads();
    cur ^= 1;
  }

  const int mh = (lane >> 4) * 8;      // C/D layout: vgpr e -> m = e + 8*(lane>=16)
  #pragma unroll
  for (int i = 0; i < 4; ++i) {
    #pragma unroll
    for (int e = 0; e < 8; ++e) {
      int m = m0 + wm * 64 + i * 16 + mh + e;
      if (!EDGE || m < M) {
        #pragma unroll
        for (int j = 0; j < 2; ++j) {
          int n = n0 + wn * 32 + j * 16 + rl;
          float v = acc[i][j][e];
          if (bias) v += bias[n];
          C[(size_t)m * ldc + n] = v;
        }
      }
    }
  }
}

// ---------------------------------------------------------------------------
// Elementwise / reduction kernels (f32 math)
// ---------------------------------------------------------------------------
__global__ void k_cvt(const float* __restrict__ s, bf16_t* __restrict__ d, long n) {
  long i = (long)blockIdx.x * 256 + threadIdx.x;
  if (i < n) d[i] = f2bf(s[i]);
}

// per-row flag = (sum |x| > 0) ? 1 : 0   (one wave per row)
__global__ void k_rowmask(const float* __restrict__ x, int cols,
                          float* __restrict__ flag, int rows) {
  int row  = blockIdx.x * 8 + (threadIdx.x >> 5);
  int lane = threadIdx.x & 31;
  if (row >= rows) return;
  const float* p = x + (size_t)row * cols;
  float s = 0.f;
  for (int c = lane; c < cols; c += 32) s += fabsf(p[c]);
  for (int o = 16; o; o >>= 1) s += __shfl_xor(s, o, 32);
  if (lane == 0) flag[row] = (s > 0.f) ? 1.f : 0.f;
}

// bv[(br,c,x),l] = bias[l] + m*(Pc[(br,c),l] + Px[(br,x),l]);  z = bf16(qe4*bv)
__global__ void k_combine(const float* __restrict__ Pc, const float* __restrict__ Px,
                          const float* __restrict__ bias,
                          const float* __restrict__ mc, const float* __restrict__ mx,
                          const float* __restrict__ qe4,
                          float* __restrict__ bv, bf16_t* __restrict__ z,
                          int nBR, int nc, int nx) {
  long total = (long)nBR * nc * nx * L_;
  long idx = (long)blockIdx.x * 256 + threadIdx.x;
  if (idx >= total) return;
  int  l = (int)(idx & (L_ - 1));
  long t = idx >> 9;
  int  x = (int)(t % nx); t /= nx;
  int  c = (int)(t % nc);
  int  br = (int)(t / nc);
  float m = mc[(long)br * nc + c] * mx[(long)br * nx + x];
  float val = bias[l] + m * (Pc[((long)br * nc + c) * L_ + l] +
                             Px[((long)br * nx + x) * L_ + l]);
  bv[idx] = val;
  z[idx]  = f2bf(qe4[(long)br * L_ + l] * val);
}

// softmax of sc over x (stride L_), then z = bf16(softmax * bv)
__global__ void k_softmax_mul(const float* __restrict__ sc, const float* __restrict__ bv,
                              bf16_t* __restrict__ z, int nBRC, int nx) {
  long total = (long)nBRC * L_;
  long idx = (long)blockIdx.x * 256 + threadIdx.x;
  if (idx >= total) return;
  int  l   = (int)(idx & (L_ - 1));
  long brc = idx >> 9;
  const float* ps = sc + (brc * nx) * (long)L_ + l;
  float mx = -INFINITY;
  for (int x = 0; x < nx; ++x) mx = fmaxf(mx, ps[(long)x * L_]);
  float s = 0.f;
  for (int x = 0; x < nx; ++x) s += expf(ps[(long)x * L_] - mx);
  float inv = 1.f / s;
  const float* pb = bv + (brc * nx) * (long)L_ + l;
  bf16_t*      pz = z  + (brc * nx) * (long)L_ + l;
  for (int x = 0; x < nx; ++x)
    pz[(long)x * L_] = f2bf(expf(ps[(long)x * L_] - mx) * inv * pb[(long)x * L_]);
}

// out[row] = sum_l wv[l]*qe[br,l]*(S[row,l] + E?[(br,x),l] + bias?[l]) + bsc[0]
__global__ void k_score(const float* __restrict__ S, const float* __restrict__ E,
                        const float* __restrict__ bias,
                        const float* __restrict__ qe, const float* __restrict__ wv,
                        const float* __restrict__ bsc, float* __restrict__ out,
                        int nBR, int nc, int nx) {
  long row  = (long)blockIdx.x * 8 + (threadIdx.x >> 5);
  int  lane = threadIdx.x & 31;
  long nrows = (long)nBR * nc * nx;
  if (row >= nrows) return;
  int  x  = (int)(row % nx);
  long br = row / ((long)nc * nx);
  const float* ps = S + row * L_;
  const float* pe = E ? (E + ((long)br * nx + x) * L_) : nullptr;
  const float* pq = qe + (long)br * L_;
  float s = 0.f;
  for (int l = lane; l < L_; l += 32) {
    float v = ps[l];
    if (pe)   v += pe[l];
    if (bias) v += bias[l];
    s += wv[l] * pq[l] * v;
  }
  for (int o = 16; o; o >>= 1) s += __shfl_xor(s, o, 32);
  if (lane == 0) out[row] = s + bsc[0];
}

// in-place softmax over contiguous last dim (one thread per group)
__global__ void k_softmax_vec(float* __restrict__ sc, int ngroups, int nx) {
  int g = blockIdx.x * 256 + threadIdx.x;
  if (g >= ngroups) return;
  float* p = sc + (long)g * nx;
  float mx = -INFINITY;
  for (int i = 0; i < nx; ++i) mx = fmaxf(mx, p[i]);
  float s = 0.f;
  for (int i = 0; i < nx; ++i) { float e = expf(p[i] - mx); p[i] = e; s += e; }
  float inv = 1.f / s;
  for (int i = 0; i < nx; ++i) p[i] *= inv;
}

// out[(br,c),l] = sum_x d[(br,c),x] * X[(br,x),l]
__global__ void k_wsum(const float* __restrict__ d, const float* __restrict__ X,
                       float* __restrict__ out, int nBR, int nc, int nx, int D) {
  long total = (long)nBR * nc * D;
  long idx = (long)blockIdx.x * 256 + threadIdx.x;
  if (idx >= total) return;
  int  l = (int)(idx % D);
  long t = idx / D;
  int  c  = (int)(t % nc);
  int  br = (int)(t / nc);
  const float* dd = d + ((long)br * nc + c) * nx;
  const float* xx = X + ((long)br * nx) * D + l;
  float s = 0.f;
  for (int x = 0; x < nx; ++x) s += dd[x] * xx[(long)x * D];
  out[idx] = s;
}

// copy src[r, 0:cols] -> dstF/dstB[r, doff:doff+cols] (f32 and/or bf16)
__global__ void k_pack(const float* __restrict__ src, int sld,
                       float* __restrict__ dstF, bf16_t* __restrict__ dstB,
                       int dld, int doff, long rows, int cols) {
  long total = rows * cols;
  long idx = (long)blockIdx.x * 256 + threadIdx.x;
  if (idx >= total) return;
  int  c = (int)(idx % cols);
  long r = idx / cols;
  float v = src[r * sld + c];
  if (dstF) dstF[r * dld + doff + c] = v;
  if (dstB) dstB[r * dld + doff + c] = f2bf(v);
}

// o = bf16(sigmoid(g) * x)
__global__ void k_glu(const float* __restrict__ g, const float* __restrict__ x,
                      bf16_t* __restrict__ o, long n) {
  long i = (long)blockIdx.x * 256 + threadIdx.x;
  if (i < n) {
    float sg = 1.f / (1.f + expf(-g[i]));
    o[i] = f2bf(sg * x[i]);
  }
}

// ---------------------------------------------------------------------------
// Host orchestration
// ---------------------------------------------------------------------------
extern "C" void kernel_launch(void* const* d_in, const int* in_sizes, int n_in,
                              void* d_out, int out_size, void* d_ws, size_t ws_size,
                              hipStream_t stream) {
  (void)in_sizes; (void)n_in; (void)out_size; (void)ws_size;

  const float* v_nodes = (const float*)d_in[0];
  const float* t_nodes = (const float*)d_in[1];
  const float* ques    = (const float*)d_in[2];

  enum {  // _LAYER_DIMS insertion order; d_in[3+2i]=w, d_in[4+2i]=b
    LW_T2V_4 = 0, LW_T2V_5, LW_T2V_B, LW_V2T_4, LW_V2T_5, LW_V2T_B,
    LW_T2V_6, LW_T2V_C, LW_V2T_6, LW_V2T_C, LW_T2V_L, LW_T2V_7,
    LW_V2T_L, LW_V2T_7, LW_T2V_E, LW_T2V_8, LW_T2V_A, LW_T2V_9,
    LW_T2V_LG, LW_T2V_NR, LW_V2T_E, LW_V2T_8, LW_V2T_A, LW_V2T_9,
    LW_V2T_LG, LW_V2T_NR
  };
  auto Wf = [&](int i) { return (const float*)d_in[3 + 2 * i]; };
  auto Bf = [&](int i) { return (const float*)d_in[4 + 2 * i]; };

  char* wsb = (char*)d_ws;
  size_t off = 0;
  auto alloc = [&](size_t bytes) -> void* {
    void* p = (void*)(wsb + off);
    off += (bytes + 255) & ~(size_t)255;
    return p;
  };
  auto g1 = [](long n) { return dim3((unsigned)((n + 255) / 256)); };
  auto gw = [](long rows) { return dim3((unsigned)((rows + 7) / 8)); };

  auto cvt = [&](const float* s, bf16_t* d, long n) {
    k_cvt<<<g1(n), dim3(256), 0, stream>>>(s, d, n);
  };
  auto mkW = [&](int li, long n) -> bf16_t* {
    bf16_t* p = (bf16_t*)alloc((size_t)n * 2);
    cvt(Wf(li), p, n);
    return p;
  };
  auto gemm = [&](const bf16_t* A, int lda, const bf16_t* Wb, int ldw,
                  const float* bias, float* C, int ldc, int M, int N, int K) {
    int mfull = M / GBM;
    if (mfull > 0) {
      dim3 g((unsigned)(N / GBN), (unsigned)mfull);
      k_gemm_wmma<false><<<g, dim3(256), 0, stream>>>(
          A, lda, Wb, ldw, bias, C, ldc, M, N, K, 0);
    }
    if (M - mfull * GBM > 0) {
      dim3 g((unsigned)(N / GBN), 1u);
      k_gemm_wmma<true><<<g, dim3(256), 0, stream>>>(
          A, lda, Wb, ldw, bias, C, ldc, M, N, K, mfull * GBM);
    }
  };

  const int  nBR   = B_ * R_;              // 80
  const long rowsV = (long)nBR * O_;       // 2880
  const long rowsT = (long)nBR * R_;       // 800
  const long rowsE = (long)nBR * O_ * R_;  // 28800

  // ---- bf16 activations & weights ----
  bf16_t* v_bf = (bf16_t*)alloc(rowsV * V_ * 2);       cvt(v_nodes, v_bf, rowsV * V_);
  bf16_t* t_bf = (bf16_t*)alloc(rowsT * L_ * 2);       cvt(t_nodes, t_bf, rowsT * L_);
  bf16_t* q_bf = (bf16_t*)alloc((long)nBR * L_ * 2);   cvt(ques, q_bf, (long)nBR * L_);

  bf16_t* wb_t2v_4  = mkW(LW_T2V_4,  512L * 512);
  bf16_t* wb_t2v_5  = mkW(LW_T2V_5,  512L * 2560);
  bf16_t* wb_t2v_b  = mkW(LW_T2V_B,  512L * 512);
  bf16_t* wb_v2t_4  = mkW(LW_V2T_4,  512L * 512);
  bf16_t* wb_v2t_5  = mkW(LW_V2T_5,  512L * 2560);
  bf16_t* wb_v2t_b  = mkW(LW_V2T_B,  512L * 512);
  bf16_t* wb_t2v_6  = mkW(LW_T2V_6,  512L * 1024);
  bf16_t* wb_v2t_6  = mkW(LW_V2T_6,  512L * 2560);
  bf16_t* wb_t2v_l  = mkW(LW_T2V_L,  2560L * 2560);
  bf16_t* wb_t2v_7  = mkW(LW_T2V_7,  512L * 2560);
  bf16_t* wb_v2t_l  = mkW(LW_V2T_L,  2560L * 2560);
  bf16_t* wb_v2t_7  = mkW(LW_V2T_7,  512L * 2560);
  bf16_t* wb_t2v_8  = mkW(LW_T2V_8,  512L * 2048);
  bf16_t* wb_t2v_9  = mkW(LW_T2V_9,  512L * 512);
  bf16_t* wb_t2v_lg = mkW(LW_T2V_LG, 2560L * 2560);
  bf16_t* wb_t2v_nr = mkW(LW_T2V_NR, 512L * 2560);
  bf16_t* wb_v2t_8  = mkW(LW_V2T_8,  512L * 512);
  bf16_t* wb_v2t_9  = mkW(LW_V2T_9,  512L * 512);
  bf16_t* wb_v2t_lg = mkW(LW_V2T_LG, 1024L * 1024);
  bf16_t* wb_v2t_nr = mkW(LW_V2T_NR, 512L * 1024);

  // ---- masks ----
  float* maskV = (float*)alloc(rowsV * 4);
  float* maskT = (float*)alloc(rowsT * 4);
  k_rowmask<<<gw(rowsV), dim3(256), 0, stream>>>(v_nodes, V_, maskV, (int)rowsV);
  k_rowmask<<<gw(rowsT), dim3(256), 0, stream>>>(t_nodes, L_, maskT, (int)rowsT);

  // ---- scratch ----
  float*  Pv5  = (float*)alloc(rowsV * L_ * 4);
  float*  Qt5  = (float*)alloc(rowsT * L_ * 4);
  float*  Pt5  = (float*)alloc(rowsT * L_ * 4);
  float*  Qv5  = (float*)alloc(rowsV * L_ * 4);
  float*  qe4t = (float*)alloc((long)nBR * L_ * 4);
  float*  qe4v = (float*)alloc((long)nBR * L_ * 4);
  float*  BIG1 = (float*)alloc(rowsE * L_ * 4);   // b_v / b_s / gating G
  float*  BIG2 = (float*)alloc(rowsE * L_ * 4);   // scores / S
  bf16_t* ZB   = (bf16_t*)alloc(rowsE * L_ * 2);  // z / updated bridge (bf16)
  float*  T6   = (float*)alloc(rowsT * L_ * 4);
  float*  V6   = (float*)alloc(rowsV * L_ * 4);
  float*  scr  = (float*)alloc(rowsE * 4);
  float*  updV = (float*)alloc(rowsV * L_ * 4);
  float*  updT = (float*)alloc(rowsT * V_ * 4);

  // ================= t2v bridge =================
  gemm(v_bf, V_, wb_t2v_5,        2560, nullptr, Pv5, L_, (int)rowsV, L_, V_);
  gemm(t_bf, L_, wb_t2v_5 + 2048, 2560, nullptr, Qt5, L_, (int)rowsT, L_, L_);
  gemm(q_bf, L_, wb_t2v_4, 512, Bf(LW_T2V_4), qe4t, L_, nBR, L_, L_);
  k_combine<<<g1(rowsE * L_), dim3(256), 0, stream>>>(
      Pv5, Qt5, Bf(LW_T2V_5), maskV, maskT, qe4t, BIG1, ZB, nBR, O_, R_);
  gemm(ZB, L_, wb_t2v_b, 512, Bf(LW_T2V_B), BIG2, L_, (int)rowsE, L_, L_);
  k_softmax_mul<<<g1(rowsV * L_), dim3(256), 0, stream>>>(BIG2, BIG1, ZB, nBR * O_, R_);
  gemm(t_bf, L_, wb_t2v_6,       1024, nullptr, T6,   L_, (int)rowsT, L_, L_);
  gemm(ZB,   L_, wb_t2v_6 + 512, 1024, nullptr, BIG2, L_, (int)rowsE, L_, L_);
  k_score<<<gw(rowsE), dim3(256), 0, stream>>>(
      BIG2, T6, Bf(LW_T2V_6), qe4t, Wf(LW_T2V_C), Bf(LW_T2V_C), scr, nBR, O_, R_);
  k_softmax_vec<<<g1(rowsV), dim3(256), 0, stream>>>(scr, nBR * O_, R_);
  k_wsum<<<g1(rowsV * L_), dim3(256), 0, stream>>>(scr, t_nodes, updV, nBR, O_, R_, L_);

  // ================= v2t bridge =================
  gemm(t_bf, L_, wb_v2t_5,       2560, nullptr, Pt5, L_, (int)rowsT, L_, L_);
  gemm(v_bf, V_, wb_v2t_5 + 512, 2560, nullptr, Qv5, L_, (int)rowsV, L_, V_);
  gemm(q_bf, L_, wb_v2t_4, 512, Bf(LW_V2T_4), qe4v, L_, nBR, L_, L_);
  k_combine<<<g1(rowsE * L_), dim3(256), 0, stream>>>(
      Pt5, Qv5, Bf(LW_V2T_5), maskT, maskV, qe4v, BIG1, ZB, nBR, R_, O_);
  gemm(ZB, L_, wb_v2t_b, 512, Bf(LW_V2T_B), BIG2, L_, (int)rowsE, L_, L_);
  k_softmax_mul<<<g1(rowsT * L_), dim3(256), 0, stream>>>(BIG2, BIG1, ZB, nBR * R_, O_);
  gemm(v_bf, V_, wb_v2t_6,        2560, nullptr, V6,   L_, (int)rowsV, L_, V_);
  gemm(ZB,   L_, wb_v2t_6 + 2048, 2560, nullptr, BIG2, L_, (int)rowsE, L_, L_);
  k_score<<<gw(rowsE), dim3(256), 0, stream>>>(
      BIG2, V6, Bf(LW_V2T_6), qe4v, Wf(LW_V2T_C), Bf(LW_V2T_C), scr, nBR, R_, O_);
  k_softmax_vec<<<g1(rowsT), dim3(256), 0, stream>>>(scr, nBR * R_, O_);
  k_wsum<<<g1(rowsT * V_), dim3(256), 0, stream>>>(scr, v_nodes, updT, nBR, R_, O_, V_);

  // ================= t2v aggregation -> I =================
  float*  catVf = (float*)alloc(rowsV * 2560 * 4);
  bf16_t* catVb = (bf16_t*)alloc(rowsV * 2560 * 2);
  k_pack<<<g1(rowsV * V_), dim3(256), 0, stream>>>(v_nodes, V_, catVf, catVb, 2560, 0,    rowsV, V_);
  k_pack<<<g1(rowsV * L_), dim3(256), 0, stream>>>(updV,    L_, catVf, catVb, 2560, 2048, rowsV, L_);
  gemm(catVb, 2560, wb_t2v_l, 2560, Bf(LW_T2V_L), BIG1, 2560, (int)rowsV, 2560, 2560);
  bf16_t* gluB = (bf16_t*)alloc(rowsV * 2560 * 2);
  k_glu<<<g1(rowsV * 2560), dim3(256), 0, stream>>>(BIG1, catVf, gluB, rowsV * 2560);
  float* localV = (float*)alloc(rowsV * L_ * 4);
  gemm(gluB, 2560, wb_t2v_7, 2560, Bf(LW_T2V_7), localV, L_, (int)rowsV, L_, 2560);
  bf16_t* localVb = (bf16_t*)alloc(rowsV * L_ * 2);
  cvt(localV, localVb, rowsV * L_);

  float* X8 = (float*)alloc(rowsV * L_ * 4);
  gemm(v_bf, V_, wb_t2v_8, 2048, Bf(LW_T2V_8), X8, L_, (int)rowsV, L_, V_);
  float* sc1 = (float*)alloc(rowsV * 4);
  k_score<<<gw(rowsV), dim3(256), 0, stream>>>(
      X8, nullptr, nullptr, ques, Wf(LW_T2V_E), Bf(LW_T2V_E), sc1, nBR, 1, O_);
  k_softmax_vec<<<g1(nBR), dim3(256), 0, stream>>>(sc1, nBR, O_);
  float* Io = (float*)alloc((long)nBR * V_ * 4);
  k_wsum<<<g1((long)nBR * V_), dim3(256), 0, stream>>>(sc1, v_nodes, Io, nBR, 1, O_, V_);

  float* X9 = (float*)alloc(rowsV * L_ * 4);
  gemm(localVb, L_, wb_t2v_9, 512, Bf(LW_T2V_9), X9, L_, (int)rowsV, L_, L_);
  float* sc2 = (float*)alloc(rowsV * 4);
  k_score<<<gw(rowsV), dim3(256), 0, stream>>>(
      X9, nullptr, nullptr, ques, Wf(LW_T2V_A), Bf(LW_T2V_A), sc2, nBR, 1, O_);
  k_softmax_vec<<<g1(nBR), dim3(256), 0, stream>>>(sc2, nBR, O_);
  float* Ic = (float*)alloc((long)nBR * L_ * 4);
  k_wsum<<<g1((long)nBR * L_), dim3(256), 0, stream>>>(sc2, localV, Ic, nBR, 1, O_, L_);

  float*  catIf = (float*)alloc((long)nBR * 2560 * 4);
  bf16_t* catIb = (bf16_t*)alloc((long)nBR * 2560 * 2);
  k_pack<<<g1((long)nBR * V_), dim3(256), 0, stream>>>(Io, V_, catIf, catIb, 2560, 0,    nBR, V_);
  k_pack<<<g1((long)nBR * L_), dim3(256), 0, stream>>>(Ic, L_, catIf, catIb, 2560, 2048, nBR, L_);
  float* G2 = (float*)alloc((long)nBR * 2560 * 4);
  gemm(catIb, 2560, wb_t2v_lg, 2560, Bf(LW_T2V_LG), G2, 2560, nBR, 2560, 2560);
  bf16_t* glu2 = (bf16_t*)alloc((long)nBR * 2560 * 2);
  k_glu<<<g1((long)nBR * 2560), dim3(256), 0, stream>>>(G2, catIf, glu2, (long)nBR * 2560);
  gemm(glu2, 2560, wb_t2v_nr, 2560, Bf(LW_T2V_NR), (float*)d_out, L_, nBR, L_, 2560);

  // ================= v2t aggregation -> H =================
  float*  catTf = (float*)alloc(rowsT * 2560 * 4);
  bf16_t* catTb = (bf16_t*)alloc(rowsT * 2560 * 2);
  k_pack<<<g1(rowsT * L_), dim3(256), 0, stream>>>(t_nodes, L_, catTf, catTb, 2560, 0,   rowsT, L_);
  k_pack<<<g1(rowsT * V_), dim3(256), 0, stream>>>(updT,    V_, catTf, catTb, 2560, 512, rowsT, V_);
  gemm(catTb, 2560, wb_v2t_l, 2560, Bf(LW_V2T_L), BIG1, 2560, (int)rowsT, 2560, 2560);
  k_glu<<<g1(rowsT * 2560), dim3(256), 0, stream>>>(BIG1, catTf, gluB, rowsT * 2560);
  float* localT = (float*)alloc(rowsT * L_ * 4);
  gemm(gluB, 2560, wb_v2t_7, 2560, Bf(LW_V2T_7), localT, L_, (int)rowsT, L_, 2560);
  bf16_t* localTb = (bf16_t*)alloc(rowsT * L_ * 2);
  cvt(localT, localTb, rowsT * L_);

  float* T8 = (float*)alloc(rowsT * L_ * 4);
  gemm(t_bf, L_, wb_v2t_8, 512, Bf(LW_V2T_8), T8, L_, (int)rowsT, L_, L_);
  float* sc3 = (float*)alloc(rowsT * 4);
  k_score<<<gw(rowsT), dim3(256), 0, stream>>>(
      T8, nullptr, nullptr, ques, Wf(LW_V2T_E), Bf(LW_V2T_E), sc3, nBR, 1, R_);
  k_softmax_vec<<<g1(nBR), dim3(256), 0, stream>>>(sc3, nBR, R_);
  float* Ho = (float*)alloc((long)nBR * L_ * 4);
  k_wsum<<<g1((long)nBR * L_), dim3(256), 0, stream>>>(sc3, t_nodes, Ho, nBR, 1, R_, L_);

  float* T9 = (float*)alloc(rowsT * L_ * 4);
  gemm(localTb, L_, wb_v2t_9, 512, Bf(LW_V2T_9), T9, L_, (int)rowsT, L_, L_);
  float* sc4 = (float*)alloc(rowsT * 4);
  k_score<<<gw(rowsT), dim3(256), 0, stream>>>(
      T9, nullptr, nullptr, ques, Wf(LW_V2T_A), Bf(LW_V2T_A), sc4, nBR, 1, R_);
  k_softmax_vec<<<g1(nBR), dim3(256), 0, stream>>>(sc4, nBR, R_);
  float* Hc = (float*)alloc((long)nBR * L_ * 4);
  k_wsum<<<g1((long)nBR * L_), dim3(256), 0, stream>>>(sc4, localT, Hc, nBR, 1, R_, L_);

  float*  catHf = (float*)alloc((long)nBR * 1024 * 4);
  bf16_t* catHb = (bf16_t*)alloc((long)nBR * 1024 * 2);
  k_pack<<<g1((long)nBR * L_), dim3(256), 0, stream>>>(Ho, L_, catHf, catHb, 1024, 0,   nBR, L_);
  k_pack<<<g1((long)nBR * L_), dim3(256), 0, stream>>>(Hc, L_, catHf, catHb, 1024, 512, nBR, L_);
  float* G4 = (float*)alloc((long)nBR * 1024 * 4);
  gemm(catHb, 1024, wb_v2t_lg, 1024, Bf(LW_V2T_LG), G4, 1024, nBR, 1024, 1024);
  bf16_t* glu4 = (bf16_t*)alloc((long)nBR * 1024 * 2);
  k_glu<<<g1((long)nBR * 1024), dim3(256), 0, stream>>>(G4, catHf, glu4, (long)nBR * 1024);
  gemm(glu4, 1024, wb_v2t_nr, 1024, Bf(LW_V2T_NR),
       (float*)d_out + (long)nBR * L_, L_, nBR, L_, 1024);
}
